// _EncoderLayer_50835232915827
// MI455X (gfx1250) — matrix operand order repaired and checked by reading, not compiled, not run on hardware
//
#include <hip/hip_runtime.h>
#include <hip/hip_bf16.h>
#include <math.h>

#define S_   1024
#define B_   8
#define D_   1024
#define H_   16
#define DH_  64
#define DFF_ 4096
#define EPS_ 1e-5f

typedef __attribute__((ext_vector_type(16))) _Float16 v16h;
typedef __attribute__((ext_vector_type(8)))  float    v8f;
typedef __attribute__((ext_vector_type(4)))  unsigned v4u;
typedef __attribute__((ext_vector_type(8)))  int      v8i;
typedef __attribute__((ext_vector_type(4)))  int      v4i;

#if defined(__has_builtin)
#if __has_builtin(__builtin_amdgcn_tensor_load_to_lds)
#define USE_TDM 1
#endif
#endif
#ifndef USE_TDM
#define USE_TDM 0
#endif

union Frag16 { v16h v; uint4 u[2]; };
union Half8  { uint4 u; _Float16 h[8]; };
union Half4  { uint2 u; _Float16 h[4]; };

__device__ __forceinline__ v8f wmma_f16(v16h a, v16h b, v8f c) {
  // D = A(16x32) * B(32x16) + C, f32 accumulate
  return __builtin_amdgcn_wmma_f32_16x16x32_f16(false, a, false, b, (short)0, c,
                                                false, false);
}

// A-matrix fragment (16x32 f16). Element (m,k) at p[m*stride + k].
// lane(0-15)=m, half i: k = (i<8 ? i : i+8) + 8*hi -> two 16B chunks.
__device__ __forceinline__ v16h ldA(const _Float16* p, int stride) {
  int lane = threadIdx.x & 31;
  int m = lane & 15, hi = lane >> 4;
  const _Float16* r = p + m * stride + 8 * hi;
  Frag16 t;
  t.u[0] = *(const uint4*)(r);
  t.u[1] = *(const uint4*)(r + 16);
  return t.v;
}

// B-matrix fragment (32x16 f16). Element (k,n) at p[n*stride + k]
// (B stored k-contiguous per n-row). lane(0-15)=n, half i: k = (i<8 ? i : i+8) + 8*hi
// -> two 16B chunks.
__device__ __forceinline__ v16h ldB(const _Float16* p, int stride) {
  int lane = threadIdx.x & 31;
  int n = lane & 15, hi = lane >> 4;
  const _Float16* r = p + n * stride + 8 * hi;
  Frag16 t;
  t.u[0] = *(const uint4*)(r);
  t.u[1] = *(const uint4*)(r + 16);
  return t.v;
}

#if USE_TDM
// TDM: DMA a (128 rows x 32 halfs) tile, row stride K halfs, into LDS with
// 16B padding after every 64B -> LDS row stride 40 halfs.
__device__ __forceinline__ void tdm_tile(unsigned lds_byte, const _Float16* g,
                                         unsigned Kdim, unsigned rows) {
  unsigned long long ga = (unsigned long long)(const void*)g;
  v4u g0;
  g0[0] = 1u;                                  // count=1, user descriptor
  g0[1] = lds_byte;                            // lds_addr (bytes)
  g0[2] = (unsigned)ga;                        // global_addr lo
  g0[3] = (unsigned)(ga >> 32) | (2u << 30);   // global_addr hi | type=2
  v8i g1;
  // data_size=2B, pad_enable, pad_interval=16 DWORD (64B), pad_amount=4 DWORD
  g1[0] = (int)((1u << 16) | (1u << 20) | (3u << 22) | (3u << 25));
  g1[1] = (int)((Kdim & 0xFFFFu) << 16);                  // tensor_dim0 lo16
  g1[2] = (int)((Kdim >> 16) | ((rows & 0xFFFFu) << 16)); // dim0 hi | dim1 lo
  g1[3] = (int)((rows >> 16) | (32u << 16));              // dim1 hi | tile_dim0=32
  g1[4] = 128;                                            // tile_dim1=128
  g1[5] = (int)Kdim;                                      // dim0_stride lo32
  g1[6] = 0;
  g1[7] = 0;
  v4i gz = {0, 0, 0, 0};
  v8i gz8 = {0, 0, 0, 0, 0, 0, 0, 0};
  __builtin_amdgcn_tensor_load_to_lds(g0, g1, gz, gz, gz8, 0);
}
#endif

// ----------------------------------------- f32 (K,N) -> f16 (N,K) transpose
__global__ __launch_bounds__(256) void cvtT(const float* __restrict__ in,
                                            _Float16* __restrict__ out,
                                            int Kdim, int Ndim) {
  __shared__ _Float16 t[32][33];
  int lx = threadIdx.x & 31, ly = threadIdx.x >> 5;  // 32 x 8
  int n0 = blockIdx.x * 32, k0 = blockIdx.y * 32;
#pragma unroll
  for (int r = 0; r < 32; r += 8)
    t[ly + r][lx] = (_Float16)in[(size_t)(k0 + ly + r) * Ndim + n0 + lx];
  __syncthreads();
#pragma unroll
  for (int r = 0; r < 32; r += 8)
    out[(size_t)(n0 + ly + r) * Kdim + k0 + lx] = t[lx][ly + r];
}

// ------------------------------------------- LayerNorm (S,B,D)->(B,S,D) f16
__global__ __launch_bounds__(256) void ln_f16(const float* __restrict__ x,
                                              const float* __restrict__ g,
                                              const float* __restrict__ be,
                                              _Float16* __restrict__ out) {
  __shared__ float red[256];
  int tid = threadIdx.x;
  int rid = blockIdx.x;              // rid = s*B_ + b
  int s = rid >> 3, b = rid & 7;     // B_ == 8
  const float* xr = x + (size_t)rid * D_;
  _Float16* orow = out + ((size_t)b * S_ + s) * D_;

  float4 xv = *(const float4*)(xr + tid * 4);
  red[tid] = xv.x + xv.y + xv.z + xv.w;
  for (int off = 128; off > 0; off >>= 1) {
    __syncthreads();
    if (tid < off) red[tid] += red[tid + off];
  }
  __syncthreads();
  float mean = red[0] * (1.0f / D_);
  __syncthreads();

  float d0 = xv.x - mean, d1 = xv.y - mean, d2 = xv.z - mean, d3 = xv.w - mean;
  red[tid] = d0 * d0 + d1 * d1 + d2 * d2 + d3 * d3;
  for (int off = 128; off > 0; off >>= 1) {
    __syncthreads();
    if (tid < off) red[tid] += red[tid + off];
  }
  __syncthreads();
  float rstd = rsqrtf(red[0] * (1.0f / D_) + EPS_);

  int c = tid * 4;
  float4 gv = *(const float4*)(g + c);
  float4 bv = *(const float4*)(be + c);
  Half4 t;
  t.h[0] = (_Float16)(d0 * rstd * gv.x + bv.x);
  t.h[1] = (_Float16)(d1 * rstd * gv.y + bv.y);
  t.h[2] = (_Float16)(d2 * rstd * gv.z + bv.z);
  t.h[3] = (_Float16)(d3 * rstd * gv.w + bv.w);
  *(uint2*)(orow + c) = t.u;
}

// --------------------------------------------------------- WMMA tiled GEMM
// C[M,N] = A[M,K](f16) * BT[N,K](f16, pre-transposed) (+bias)(relu?)(*scale)
// mode 0: out16[m*N+n] = f16 result
// mode 1: out32[(s*B_+b)*N+n] = result + resid[...], where m = b*S_+s
__global__ __launch_bounds__(256) void gemm_wmma(
    const _Float16* __restrict__ A, const _Float16* __restrict__ BT,
    const float* __restrict__ bias, const float* __restrict__ resid,
    _Float16* __restrict__ out16, float* __restrict__ out32,
    int M, int N, int K, float scale, int relu, int mode) {
  __shared__ _Float16 sm[2][2][128 * 40];  // [buf][A/B][row*40 + k]

  int tid = threadIdx.x, lane = tid & 31, wid = tid >> 5;
  int wm = wid >> 2, wn = wid & 3;          // 2 x 4 wave grid
  int ctaM = blockIdx.y * 128, ctaN = blockIdx.x * 128;
  int nk = K >> 5;

  v8f acc[4][2] = {};

#if USE_TDM
  if (wid == 0) {  // one wave drives the Tensor Data Mover
    tdm_tile((unsigned)(unsigned long long)(const void*)&sm[0][0][0],
             A + (size_t)ctaM * K, K, M);
    tdm_tile((unsigned)(unsigned long long)(const void*)&sm[0][1][0],
             BT + (size_t)ctaN * K, K, N);
  }
#endif

  for (int kt = 0; kt < nk; ++kt) {
    int cur = kt & 1;
#if USE_TDM
    if (wid == 0) {
      if (kt + 1 < nk) {
        int k0n = (kt + 1) << 5;
        tdm_tile((unsigned)(unsigned long long)(const void*)&sm[cur ^ 1][0][0],
                 A + (size_t)ctaM * K + k0n, K, M);
        tdm_tile((unsigned)(unsigned long long)(const void*)&sm[cur ^ 1][1][0],
                 BT + (size_t)ctaN * K + k0n, K, N);
        __builtin_amdgcn_s_wait_tensorcnt(2);  // current buffer complete
      } else {
        __builtin_amdgcn_s_wait_tensorcnt(0);
      }
    }
    __syncthreads();
#else
    int k0 = kt << 5;
    int c = tid * 2;
    for (int it = 0; it < 2; ++it, ++c) {
      int row = c >> 2, col = (c & 3) * 8;
      *(uint4*)(&sm[cur][0][row * 40 + col]) =
          *(const uint4*)(A + (size_t)(ctaM + row) * K + k0 + col);
      *(uint4*)(&sm[cur][1][row * 40 + col]) =
          *(const uint4*)(BT + (size_t)(ctaN + row) * K + k0 + col);
    }
    __syncthreads();
#endif

    v16h af[4], bf[2];
#pragma unroll
    for (int mi = 0; mi < 4; ++mi)
      af[mi] = ldA(&sm[cur][0][(wm * 64 + mi * 16) * 40], 40);
#pragma unroll
    for (int ni = 0; ni < 2; ++ni)
      bf[ni] = ldB(&sm[cur][1][(wn * 32 + ni * 16) * 40], 40);
#pragma unroll
    for (int mi = 0; mi < 4; ++mi)
#pragma unroll
      for (int ni = 0; ni < 2; ++ni)
        acc[mi][ni] = wmma_f16(af[mi], bf[ni], acc[mi][ni]);
    __syncthreads();
  }

  int nl = lane & 15, hi = lane >> 4;
#pragma unroll
  for (int mi = 0; mi < 4; ++mi)
#pragma unroll
    for (int ni = 0; ni < 2; ++ni) {
      int n = ctaN + wn * 32 + ni * 16 + nl;
      float bn = bias[n];
#pragma unroll
      for (int g = 0; g < 8; ++g) {
        int m = ctaM + wm * 64 + mi * 16 + g + 8 * hi;
        float v = acc[mi][ni][g] + bn;
        if (relu) v = fmaxf(v, 0.0f);
        v *= scale;
        if (mode == 0) {
          out16[(size_t)m * N + n] = (_Float16)v;
        } else {
          int bb = m >> 10, ss = m & 1023;              // S_ == 1024
          size_t idx = ((size_t)ss * B_ + bb) * N + n;  // (S,B,D) layout
          out32[idx] = v + resid[idx];
        }
      }
    }
}

// ------------------------------------------------ flash attention (per b,h)
// Q pre-scaled by DH^-0.5. scores += attn_bias[b,key]*alpha; mask -> -inf.
__global__ __launch_bounds__(128) void attn_flash(
    const _Float16* __restrict__ Qh, const _Float16* __restrict__ Kh,
    const _Float16* __restrict__ Vh, const float* __restrict__ abias,
    const unsigned char* __restrict__ mask, const float* __restrict__ alphap,
    _Float16* __restrict__ Oh) {
  __shared__ _Float16 Vt[64 * 72];      // [dh][key], stride 72 (V transposed)
  __shared__ float    Sl[4][16 * 68];   // per-wave scores [row][key], stride 68
  __shared__ _Float16 Pl[4][16 * 72];   // per-wave P      [row][key], stride 72
  __shared__ float    cst[4][16];       // per-wave per-row corr / l

  int tid = threadIdx.x, lane = tid & 31, wid = tid >> 5;
  int nl = lane & 15, hi = lane >> 4;
  int bh = blockIdx.y;
  int b = bh >> 4, h = bh & 15;         // H_ == 16
  int q0 = blockIdx.x * 64 + wid * 16;
  float alpha = alphap[0];

  const _Float16* qbase = Qh + ((size_t)b * S_ + q0) * D_ + h * DH_;
  v16h qa0 = ldA(qbase, D_);
  v16h qa1 = ldA(qbase + 32, D_);

  v8f o[4] = {};
  float m_run = -INFINITY, l_run = 0.0f;  // valid on lanes 0..15

  for (int j = 0; j < S_ / 64; ++j) {
    __syncthreads();
    // load V block transposed into LDS (512 8-half chunks)
#pragma unroll
    for (int it = 0; it < 4; ++it) {
      int c = tid * 4 + it;
      int key = c >> 3, d8 = (c & 7) * 8;
      Half8 t;
      t.u = *(const uint4*)(Vh + ((size_t)b * S_ + j * 64 + key) * D_ +
                            h * DH_ + d8);
#pragma unroll
      for (int u = 0; u < 8; ++u) Vt[(d8 + u) * 72 + key] = t.h[u];
    }
    __syncthreads();

    // scores S = Q * K^T; K^T fragments load straight from global (B,S,D)
    const _Float16* kbase = Kh + ((size_t)b * S_ + j * 64) * D_ + h * DH_;
#pragma unroll
    for (int t = 0; t < 4; ++t) {
      v16h kb0 = ldB(kbase + (size_t)(t * 16) * D_, D_);
      v16h kb1 = ldB(kbase + (size_t)(t * 16) * D_ + 32, D_);
      v8f z = {};
      v8f s = wmma_f16(qa0, kb0, z);
      s = wmma_f16(qa1, kb1, s);
      int key = j * 64 + t * 16 + nl;           // uniform per lane over g
      float ab = abias[b * S_ + key] * alpha;
      bool pad = mask[b * S_ + key] != 0;
#pragma unroll
      for (int g = 0; g < 8; ++g) {
        float sv = pad ? -INFINITY : (s[g] + ab);
        Sl[wid][(g + 8 * hi) * 68 + t * 16 + nl] = sv;
      }
    }
    __syncthreads();

    // online softmax: lane r owns row r
    if (lane < 16) {
      float* sr = &Sl[wid][lane * 68];
      float mx = -INFINITY;
#pragma unroll 8
      for (int c = 0; c < 64; ++c) mx = fmaxf(mx, sr[c]);
      float mn = fmaxf(m_run, mx);
      float corr = (m_run == mn) ? 1.0f : __expf(m_run - mn);
      float sum = 0.0f;
      _Float16* pr = &Pl[wid][lane * 72];
#pragma unroll 8
      for (int c = 0; c < 64; ++c) {
        float sv = sr[c];
        float e = (sv == -INFINITY) ? 0.0f : __expf(sv - mn);
        sum += e;
        pr[c] = (_Float16)e;
      }
      l_run = l_run * corr + sum;
      m_run = mn;
      cst[wid][lane] = corr;
    }
    __syncthreads();

    // rescale O by per-row corr
    float4 ca = *(float4*)&cst[wid][8 * hi];
    float4 cb = *(float4*)&cst[wid][8 * hi + 4];
    float cf[8] = {ca.x, ca.y, ca.z, ca.w, cb.x, cb.y, cb.z, cb.w};
#pragma unroll
    for (int t = 0; t < 4; ++t)
#pragma unroll
      for (int g = 0; g < 8; ++g) o[t][g] *= cf[g];

    // O += P * V
    v16h pa0 = ldA(&Pl[wid][0], 72);
    v16h pa1 = ldA(&Pl[wid][32], 72);
#pragma unroll
    for (int t = 0; t < 4; ++t) {
      v16h vb0 = ldB(&Vt[(t * 16) * 72], 72);
      v16h vb1 = ldB(&Vt[(t * 16) * 72 + 32], 72);
      o[t] = wmma_f16(pa0, vb0, o[t]);
      o[t] = wmma_f16(pa1, vb1, o[t]);
    }
  }

  if (lane < 16) cst[wid][lane] = l_run;
  __syncthreads();
  float4 la = *(float4*)&cst[wid][8 * hi];
  float4 lb = *(float4*)&cst[wid][8 * hi + 4];
  float li[8] = {1.0f / la.x, 1.0f / la.y, 1.0f / la.z, 1.0f / la.w,
                 1.0f / lb.x, 1.0f / lb.y, 1.0f / lb.z, 1.0f / lb.w};
#pragma unroll
  for (int t = 0; t < 4; ++t)
#pragma unroll
    for (int g = 0; g < 8; ++g) {
      int q = q0 + g + 8 * hi;
      int col = h * DH_ + t * 16 + nl;
      Oh[((size_t)b * S_ + q) * D_ + col] = (_Float16)(o[t][g] * li[g]);
    }
}

// ---------------------------------------------------------------- launcher
extern "C" void kernel_launch(void* const* d_in, const int* in_sizes, int n_in,
                              void* d_out, int out_size, void* d_ws, size_t ws_size,
                              hipStream_t stream) {
  const float* x     = (const float*)d_in[0];
  const unsigned char* mask = (const unsigned char*)d_in[1];  // jnp bool = u8
  const float* abias = (const float*)d_in[2];
  const float* Wq = (const float*)d_in[3];  const float* bq = (const float*)d_in[4];
  const float* Wk = (const float*)d_in[5];  const float* bk = (const float*)d_in[6];
  const float* Wv = (const float*)d_in[7];  const float* bv = (const float*)d_in[8];
  const float* Wo = (const float*)d_in[9];  const float* bo = (const float*)d_in[10];
  const float* W1 = (const float*)d_in[11]; const float* pb1 = (const float*)d_in[12];
  const float* W2 = (const float*)d_in[13]; const float* pb2 = (const float*)d_in[14];
  const float* g1 = (const float*)d_in[15]; const float* be1 = (const float*)d_in[16];
  const float* g2 = (const float*)d_in[17]; const float* be2 = (const float*)d_in[18];
  const float* alpha = (const float*)d_in[19];

  char* ws = (char*)d_ws;
  const size_t MB = 1024 * 1024;
  _Float16* wqT = (_Float16*)(ws + 0 * MB);   // (N,K) f16 transposed weights
  _Float16* wkT = (_Float16*)(ws + 2 * MB);
  _Float16* wvT = (_Float16*)(ws + 4 * MB);
  _Float16* woT = (_Float16*)(ws + 6 * MB);
  _Float16* w1T = (_Float16*)(ws + 8 * MB);
  _Float16* w2T = (_Float16*)(ws + 16 * MB);
  _Float16* xh  = (_Float16*)(ws + 24 * MB);  // LN1 out; reused as attn_out
  _Float16* Qh  = (_Float16*)(ws + 40 * MB);  // reused as LN2 out
  _Float16* Kh  = (_Float16*)(ws + 56 * MB);
  _Float16* Vh  = (_Float16*)(ws + 72 * MB);
  float*    x1  = (float*)   (ws + 88 * MB);  // residual stream (S,B,D) f32
  _Float16* h1  = (_Float16*)(ws + 120 * MB); // FFN hidden f16

  // weight convert + transpose: (K,N) f32 -> (N,K) f16
  cvtT<<<dim3(D_ / 32, D_ / 32), 256, 0, stream>>>(Wq, wqT, D_, D_);
  cvtT<<<dim3(D_ / 32, D_ / 32), 256, 0, stream>>>(Wk, wkT, D_, D_);
  cvtT<<<dim3(D_ / 32, D_ / 32), 256, 0, stream>>>(Wv, wvT, D_, D_);
  cvtT<<<dim3(D_ / 32, D_ / 32), 256, 0, stream>>>(Wo, woT, D_, D_);
  cvtT<<<dim3(DFF_ / 32, D_ / 32), 256, 0, stream>>>(W1, w1T, D_, DFF_);
  cvtT<<<dim3(D_ / 32, DFF_ / 32), 256, 0, stream>>>(W2, w2T, DFF_, D_);

  // LN1: (S,B,D) f32 -> (B,S,D) f16
  ln_f16<<<S_ * B_, 256, 0, stream>>>(x, g1, be1, xh);

  // QKV projections (scale DH^-0.5 folded into Q)
  dim3 gP(D_ / 128, (B_ * S_) / 128);  // (8, 64)
  gemm_wmma<<<gP, 256, 0, stream>>>(xh, wqT, bq, nullptr, Qh, nullptr,
                                    B_ * S_, D_, D_, 0.125f, 0, 0);
  gemm_wmma<<<gP, 256, 0, stream>>>(xh, wkT, bk, nullptr, Kh, nullptr,
                                    B_ * S_, D_, D_, 1.0f, 0, 0);
  gemm_wmma<<<gP, 256, 0, stream>>>(xh, wvT, bv, nullptr, Vh, nullptr,
                                    B_ * S_, D_, D_, 1.0f, 0, 0);

  // flash attention -> attn_out (reuses xh)
  attn_flash<<<dim3(S_ / 64, B_ * H_), 128, 0, stream>>>(Qh, Kh, Vh, abias,
                                                         mask, alpha, xh);

  // out projection + residual: x1 = x + attn_out @ Wo + bo  (S,B,D) f32
  gemm_wmma<<<gP, 256, 0, stream>>>(xh, woT, bo, x, nullptr, x1,
                                    B_ * S_, D_, D_, 1.0f, 0, 1);

  // LN2 -> f16 (reuses Qh)
  ln_f16<<<S_ * B_, 256, 0, stream>>>(x1, g2, be2, Qh);

  // FFN1 with ReLU: h1 = relu(h @ W1 + b1), f16
  dim3 gF1(DFF_ / 128, (B_ * S_) / 128);  // (32, 64)
  gemm_wmma<<<gF1, 256, 0, stream>>>(Qh, w1T, pb1, nullptr, h1, nullptr,
                                     B_ * S_, DFF_, D_, 1.0f, 1, 0);

  // FFN2 + residual -> d_out (S,B,D) f32
  gemm_wmma<<<gP, 256, 0, stream>>>(h1, w2T, pb2, x1, nullptr, (float*)d_out,
                                    B_ * S_, D_, DFF_, 1.0f, 0, 1);

  (void)in_sizes; (void)n_in; (void)out_size; (void)ws_size;
}
